// TopKDecoder_49125835932149
// MI455X (gfx1250) — compile-verified, matrix operand order checked by reading, not run on hardware
//
#include <hip/hip_runtime.h>

// ---------------- types ----------------
typedef __bf16 bf16_t;
typedef bf16_t v16bf __attribute__((ext_vector_type(16)));
typedef bf16_t v8bf  __attribute__((ext_vector_type(8)));
typedef float  v8f   __attribute__((ext_vector_type(8)));

#define B_  8
#define K_  5
#define BK  40
#define V_  32000
#define T_  48
#define H_  1024
#define EMB_ 512
#define S_  196
#define KX  2560          // EMB + H + H (concat input for fused gates GEMM)
#define G4  4096          // 4*H
#define NEG_INF (-1e30f)

__device__ __forceinline__ bf16_t to_bf16(float f) {
  unsigned int u = __float_as_uint(f);
  u += 0x7FFFu + ((u >> 16) & 1u);            // round-to-nearest-even
  unsigned short s = (unsigned short)(u >> 16);
  return __builtin_bit_cast(bf16_t, s);
}
__device__ __forceinline__ float sigmoidf_(float x) { return 1.0f / (1.0f + __expf(-x)); }

// ---------------- one-time weight prep ----------------
// W_out [H,V] fp32 -> WoutT [V,H] bf16 (N-major so WMMA B frags are contiguous)
__global__ void k_transpose_wout(const float* __restrict__ Wout, bf16_t* __restrict__ WoutT) {
  __shared__ float tile[32][33];
  int v0 = blockIdx.x * 32, h0 = blockIdx.y * 32;
  int tx = threadIdx.x, ty = threadIdx.y;
  for (int i = 0; i < 32; i += 8)
    tile[ty + i][tx] = Wout[(size_t)(h0 + ty + i) * V_ + (v0 + tx)];
  __syncthreads();
  for (int i = 0; i < 32; i += 8)
    WoutT[(size_t)(v0 + ty + i) * H_ + (h0 + tx)] = to_bf16(tile[tx][ty + i]);
}

// Wg = rows[0:1536)=W_ih, [1536:2560)=W_hh, shape [KX,G4] -> WgT [G4,KX] bf16
__global__ void k_transpose_wg(const float* __restrict__ Wih, const float* __restrict__ Whh,
                               bf16_t* __restrict__ WgT) {
  __shared__ float tile[32][33];
  int k0 = blockIdx.x * 32, n0 = blockIdx.y * 32;
  int tx = threadIdx.x, ty = threadIdx.y;
  for (int i = 0; i < 32; i += 8) {
    int k = k0 + ty + i;
    float v = (k < 1536) ? Wih[(size_t)k * G4 + (n0 + tx)]
                         : Whh[(size_t)(k - 1536) * G4 + (n0 + tx)];
    tile[ty + i][tx] = v;
  }
  __syncthreads();
  for (int i = 0; i < 32; i += 8)
    WgT[(size_t)(n0 + ty + i) * KX + (k0 + tx)] = to_bf16(tile[tx][ty + i]);
}

// ---------------- init ----------------
__global__ void k_init_state(const float* __restrict__ h0, const float* __restrict__ c0,
                             float* __restrict__ h_cur, float* __restrict__ c_cur) {
  int bk = blockIdx.x; int i = blockIdx.y * 256 + threadIdx.x;
  h_cur[bk * H_ + i] = h0[(bk / K_) * H_ + i];
  c_cur[bk * H_ + i] = c0[(bk / K_) * H_ + i];
}
__global__ void k_init_bf_zero(bf16_t* __restrict__ hbf, bf16_t* __restrict__ xfull) {
  int idx = blockIdx.x * 256 + threadIdx.x;
  if (idx < 48 * H_) hbf[idx] = to_bf16(0.f);
  if (idx < 48 * KX) xfull[idx] = to_bf16(0.f);
}
__global__ void k_init_beam(float* __restrict__ seqsc, int* __restrict__ tok_cur) {
  int bk = threadIdx.x;
  if (bk < BK) { seqsc[bk] = (bk % K_ == 0) ? 0.f : NEG_INF; tok_cur[bk] = 1; /*SOS*/ }
}

// ---------------- attention (dot attn + softmax + context) ----------------
__global__ void k_attn(const float* __restrict__ enc, const float* __restrict__ h,
                       float* __restrict__ ctx) {
  int bk = blockIdx.x, b = bk / K_, tid = threadIdx.x;
  __shared__ float sh_h[H_];
  __shared__ float sc[S_ + 4];
  __shared__ float s_max, s_inv;
  const float* encB = enc + (size_t)b * S_ * H_;
  const float* hrow = h + (size_t)bk * H_;
  for (int i = tid; i < H_; i += 256) sh_h[i] = hrow[i];
  __syncthreads();
  int wave = tid >> 5, lane = tid & 31;
  for (int s = wave; s < S_; s += 8) {
    const float* e = encB + (size_t)s * H_;
    float sum = 0.f;
    for (int i = lane; i < H_; i += 32) sum += e[i] * sh_h[i];
    for (int m = 16; m > 0; m >>= 1) sum += __shfl_xor(sum, m, 32);
    if (lane == 0) sc[s] = sum;
  }
  __syncthreads();
  if (wave == 0) {
    float m = NEG_INF;
    for (int i = lane; i < S_; i += 32) m = fmaxf(m, sc[i]);
    for (int t = 16; t > 0; t >>= 1) m = fmaxf(m, __shfl_xor(m, t, 32));
    if (lane == 0) s_max = m;
  }
  __syncthreads();
  if (wave == 0) {
    float sum = 0.f;
    for (int i = lane; i < S_; i += 32) { float e = __expf(sc[i] - s_max); sc[i] = e; sum += e; }
    for (int t = 16; t > 0; t >>= 1) sum += __shfl_xor(sum, t, 32);
    if (lane == 0) s_inv = 1.0f / sum;
  }
  __syncthreads();
  float inv = s_inv;
  for (int i = tid; i < H_; i += 256) {
    float acc = 0.f;
    for (int s = 0; s < S_; ++s) acc += sc[s] * encB[(size_t)s * H_ + i];
    ctx[(size_t)bk * H_ + i] = acc * inv;
  }
}

// ---------------- build fused GEMM input x = [embed(tok) | ctx | h] in bf16 ----------------
__global__ void k_buildx(const float* __restrict__ embed, const int* __restrict__ tok_cur,
                         const float* __restrict__ ctx, const float* __restrict__ h_cur,
                         bf16_t* __restrict__ xfull) {
  int bk = blockIdx.x; int j = blockIdx.y * 256 + threadIdx.x;
  float v;
  if (j < EMB_)           v = embed[(size_t)tok_cur[bk] * EMB_ + j];
  else if (j < EMB_ + H_) v = ctx[bk * H_ + (j - EMB_)];
  else                    v = h_cur[bk * H_ + (j - EMB_ - H_)];
  xfull[(size_t)bk * KX + j] = to_bf16(v);
}

// ---------------- WMMA GEMM: C[40,N] = A_bf16[48,K] * Bt_bf16[N,K]^T + bias ----------------
// One wave32 per n-tile; all 3 m-tiles in registers -> each B fragment loaded ONCE and
// reused for 3 independent v_wmma_f32_16x16x32_bf16 (breaks acc RAW chain, fills XDL pipe).
__global__ void k_wmma_gemm(const bf16_t* __restrict__ A, const bf16_t* __restrict__ Bt,
                            const float* __restrict__ bias, float* __restrict__ C,
                            int K, int N) {
  int tid = threadIdx.x;
  int lane = tid & 31;
  int ntile = blockIdx.x * (blockDim.x >> 5) + (tid >> 5);
  if (ntile >= (N >> 4)) return;
  int l15 = lane & 15;
  int hi = (lane >> 4) << 3;                       // 0 for lanes 0-15, 8 for 16-31
  const bf16_t* A0 = A + (size_t)(l15)      * K;   // m-tile 0: rows 0..15
  const bf16_t* A1 = A + (size_t)(16 + l15) * K;   // m-tile 1: rows 16..31
  const bf16_t* A2 = A + (size_t)(32 + l15) * K;   // m-tile 2: rows 32..47 (40..47 zero-pad)
  const bf16_t* Brow = Bt + (size_t)(ntile * 16 + l15) * K;
  v8f acc0 = {}, acc1 = {}, acc2 = {};
  for (int k = 0; k < K; k += 32) {
    v8bf bl  = *(const v8bf*)(Brow + k + hi);       // 16B loads, contiguous frags
    v8bf bh  = *(const v8bf*)(Brow + k + 16 + hi);
    v8bf a0l = *(const v8bf*)(A0 + k + hi);
    v8bf a0h = *(const v8bf*)(A0 + k + 16 + hi);
    v8bf a1l = *(const v8bf*)(A1 + k + hi);
    v8bf a1h = *(const v8bf*)(A1 + k + 16 + hi);
    v8bf a2l = *(const v8bf*)(A2 + k + hi);
    v8bf a2h = *(const v8bf*)(A2 + k + 16 + hi);
    v16bf b, a0, a1, a2;
#pragma unroll
    for (int i = 0; i < 8; ++i) {
      b[i]  = bl[i];  b[i + 8]  = bh[i];
      a0[i] = a0l[i]; a0[i + 8] = a0h[i];
      a1[i] = a1l[i]; a1[i + 8] = a1h[i];
      a2[i] = a2l[i]; a2[i + 8] = a2h[i];
    }
    acc0 = __builtin_amdgcn_wmma_f32_16x16x32_bf16(false, a0, false, b, (short)0, acc0, false, false);
    acc1 = __builtin_amdgcn_wmma_f32_16x16x32_bf16(false, a1, false, b, (short)0, acc1, false, false);
    acc2 = __builtin_amdgcn_wmma_f32_16x16x32_bf16(false, a2, false, b, (short)0, acc2, false, false);
  }
  int n = ntile * 16 + l15;
  float bv = bias[n];
#pragma unroll
  for (int r = 0; r < 8; ++r) {                     // C/D layout: VGPR r -> row r (+8 hi lanes)
    int m0 = hi + r;
    C[(size_t)m0 * N + n] = acc0[r] + bv;           // rows 0..15 always valid
    C[(size_t)(16 + m0) * N + n] = acc1[r] + bv;    // rows 16..31 always valid
    int m2 = 32 + m0;
    if (m2 < BK) C[(size_t)m2 * N + n] = acc2[r] + bv;  // rows 32..39 only
  }
}

// ---------------- LSTM cell ----------------
__global__ void k_lstm(const float* __restrict__ gates, const float* __restrict__ c_cur,
                       float* __restrict__ h_tmp, float* __restrict__ c_tmp,
                       bf16_t* __restrict__ hbf) {
  int bk = blockIdx.x; int i = blockIdx.y * 256 + threadIdx.x;
  const float* g = gates + (size_t)bk * G4;
  float ig = sigmoidf_(g[i]);
  float fg = sigmoidf_(g[H_ + i]);
  float gg = tanhf(g[2 * H_ + i]);
  float og = sigmoidf_(g[3 * H_ + i]);
  float c = fg * c_cur[bk * H_ + i] + ig * gg;
  float h = og * tanhf(c);
  c_tmp[bk * H_ + i] = c;
  h_tmp[bk * H_ + i] = h;
  hbf[bk * H_ + i] = to_bf16(h);
}

// ---------------- log_softmax stats ----------------
__global__ void k_lse(const float* __restrict__ logits, float* __restrict__ rowmax,
                      float* __restrict__ lse) {
  int bk = blockIdx.x, tid = threadIdx.x;
  const float* row = logits + (size_t)bk * V_;
  __shared__ float red[256];
  float m = NEG_INF;
  for (int v = tid; v < V_; v += 256) m = fmaxf(m, row[v]);
  red[tid] = m; __syncthreads();
  for (int s = 128; s > 0; s >>= 1) { if (tid < s) red[tid] = fmaxf(red[tid], red[tid + s]); __syncthreads(); }
  float mx = red[0]; __syncthreads();
  float sum = 0.f;
  for (int v = tid; v < V_; v += 256) sum += __expf(row[v] - mx);
  red[tid] = sum; __syncthreads();
  for (int s = 128; s > 0; s >>= 1) { if (tid < s) red[tid] += red[tid + s]; __syncthreads(); }
  if (tid == 0) { rowmax[bk] = mx; lse[bk] = logf(red[0]); }
}

__global__ void k_logp(const float* __restrict__ logits, const float* __restrict__ rowmax,
                       const float* __restrict__ lse, float* __restrict__ outs, int t) {
  int bk = blockIdx.x; int v = blockIdx.y * 256 + threadIdx.x;
  float lp = logits[(size_t)bk * V_ + v] - rowmax[bk] - lse[bk];
  outs[((size_t)t * BK + bk) * V_ + v] = lp;
}

// ---------------- top-5 of K*V per batch ----------------
__global__ void k_topk(const float* __restrict__ logp_t, const float* __restrict__ seqsc,
                       int* __restrict__ cand_out, float* __restrict__ score_out) {
  int b = blockIdx.x, tid = threadIdx.x;            // 512 threads
  float lv[5]; int li[5];
#pragma unroll
  for (int i = 0; i < 5; ++i) { lv[i] = -3e38f; li[i] = 0; }
  for (int c = tid; c < K_ * V_; c += 512) {
    int k = c / V_, v = c - k * V_;
    int beam = b * K_ + k;
    float val = seqsc[beam] + logp_t[(size_t)beam * V_ + v];
    if (val > lv[4]) {
      int p = 4;
      while (p > 0 && val > lv[p - 1]) { lv[p] = lv[p - 1]; li[p] = li[p - 1]; --p; }
      lv[p] = val; li[p] = c;
    }
  }
  __shared__ float sv[512 * 5];
  __shared__ int   si[512 * 5];
#pragma unroll
  for (int i = 0; i < 5; ++i) { sv[tid * 5 + i] = lv[i]; si[tid * 5 + i] = li[i]; }
  __syncthreads();
  for (int stride = 256; stride > 0; stride >>= 1) {
    if (tid < stride) {
      float* va = &sv[tid * 5];               int* ia = &si[tid * 5];
      float* vb = &sv[(tid + stride) * 5];    int* ib = &si[(tid + stride) * 5];
      float ov[5]; int oi[5]; int p = 0, q = 0;
#pragma unroll
      for (int r = 0; r < 5; ++r) {
        if (va[p] >= vb[q]) { ov[r] = va[p]; oi[r] = ia[p]; ++p; }
        else                { ov[r] = vb[q]; oi[r] = ib[q]; ++q; }
      }
#pragma unroll
      for (int r = 0; r < 5; ++r) { va[r] = ov[r]; ia[r] = oi[r]; }
    }
    __syncthreads();
  }
  if (tid < 5) { cand_out[b * K_ + tid] = si[tid]; score_out[b * K_ + tid] = sv[tid]; }
}

// ---------------- beam bookkeeping ----------------
__global__ void k_update(const int* __restrict__ cand, const float* __restrict__ score, int t,
                         float* __restrict__ seqsc, int* __restrict__ tok_cur,
                         int* __restrict__ pred_cur, int* __restrict__ preds_ws,
                         int* __restrict__ toks_ws, float* __restrict__ stored_ws) {
  int bk = threadIdx.x;
  if (bk >= BK) return;
  int b = bk / K_;
  int c = cand[bk];
  float s = score[bk];
  int tok = c % V_;
  int pred = c / V_ + b * K_;
  preds_ws[t * BK + bk] = pred;
  toks_ws[t * BK + bk] = tok;
  stored_ws[t * BK + bk] = s;
  tok_cur[bk] = tok;
  pred_cur[bk] = pred;
  seqsc[bk] = (tok == 2 /*EOS*/) ? NEG_INF : s;
}

__global__ void k_gather_hc(const float* __restrict__ h_tmp, const float* __restrict__ c_tmp,
                            const int* __restrict__ pred_cur, float* __restrict__ h_cur,
                            float* __restrict__ c_cur) {
  int bk = blockIdx.x; int i = blockIdx.y * 256 + threadIdx.x;
  int p = pred_cur[bk];
  h_cur[bk * H_ + i] = h_tmp[p * H_ + i];
  c_cur[bk * H_ + i] = c_tmp[p * H_ + i];
}

// ---------------- finalize: fin_score + syms + tpchain ----------------
__global__ void k_backtrack(const int* __restrict__ preds_ws, const int* __restrict__ toks_ws,
                            const float* __restrict__ stored_ws, int* __restrict__ tpchain,
                            float* __restrict__ out) {
  int bk = threadIdx.x;
  if (bk >= BK) return;
  const size_t OUTS = (size_t)T_ * BK * V_;
  const size_t SYMS = OUTS;                    // syms region start
  const size_t FIN  = OUTS + (size_t)BK * T_;  // fin_score start
  // stored[T-1] rows are already sorted desc per batch -> final top_k is identity.
  out[FIN + bk] = stored_ws[(T_ - 1) * BK + bk];
  int tp = bk;
  for (int t = T_ - 1; t >= 0; --t) {
    tpchain[t * BK + bk] = tp;
    out[SYMS + (size_t)bk * T_ + t] = (float)toks_ws[t * BK + tp];
    tp = preds_ws[t * BK + tp];
  }
}

__global__ void k_reorder(const float* __restrict__ stage, const int* __restrict__ tpchain,
                          float* __restrict__ outs, int t) {
  int bk = blockIdx.x; int v = blockIdx.y * 256 + threadIdx.x;
  int src = tpchain[t * BK + bk];
  outs[((size_t)t * BK + bk) * V_ + v] = stage[(size_t)src * V_ + v];
}

__global__ void k_writehn(const float* __restrict__ h_cur, float* __restrict__ out) {
  int bk = blockIdx.x; int i = blockIdx.y * 256 + threadIdx.x;
  const size_t off = (size_t)T_ * BK * V_ + (size_t)BK * T_ + BK;
  out[off + (size_t)bk * H_ + i] = h_cur[bk * H_ + i];
}

// ---------------- host ----------------
extern "C" void kernel_launch(void* const* d_in, const int* in_sizes, int n_in,
                              void* d_out, int out_size, void* d_ws, size_t ws_size,
                              hipStream_t stream) {
  (void)in_sizes; (void)n_in; (void)out_size; (void)ws_size;
  const float* enc    = (const float*)d_in[0];
  const float* h0     = (const float*)d_in[1];
  const float* c0     = (const float*)d_in[2];
  const float* embed  = (const float*)d_in[3];
  const float* W_ih   = (const float*)d_in[4];
  const float* W_hh   = (const float*)d_in[5];
  const float* b_lstm = (const float*)d_in[6];
  const float* W_out  = (const float*)d_in[7];
  const float* b_out  = (const float*)d_in[8];
  float* out = (float*)d_out;

  char* ws = (char*)d_ws;
  size_t off = 0;
  auto alloc = [&](size_t bytes) -> void* {
    void* p = ws + off;
    off = (off + bytes + 255) & ~(size_t)255;
    return p;
  };
  bf16_t* WoutT  = (bf16_t*)alloc((size_t)V_ * H_ * 2);
  bf16_t* WgT    = (bf16_t*)alloc((size_t)G4 * KX * 2);
  bf16_t* xfull  = (bf16_t*)alloc((size_t)48 * KX * 2);
  bf16_t* hbf    = (bf16_t*)alloc((size_t)48 * H_ * 2);
  float* h_cur   = (float*)alloc((size_t)BK * H_ * 4);
  float* c_cur   = (float*)alloc((size_t)BK * H_ * 4);
  float* h_tmp   = (float*)alloc((size_t)BK * H_ * 4);
  float* c_tmp   = (float*)alloc((size_t)BK * H_ * 4);
  float* ctx     = (float*)alloc((size_t)BK * H_ * 4);
  float* gates   = (float*)alloc((size_t)BK * G4 * 4);
  float* logits  = (float*)alloc((size_t)BK * V_ * 4);
  float* stage   = (float*)alloc((size_t)BK * V_ * 4);
  float* rowmax  = (float*)alloc(256);
  float* lse     = (float*)alloc(256);
  float* seqsc   = (float*)alloc(256);
  int* tok_cur   = (int*)alloc(256);
  int* pred_cur  = (int*)alloc(256);
  int* cand_ws   = (int*)alloc(256);
  float* score_ws= (float*)alloc(256);
  int* preds_ws  = (int*)alloc((size_t)T_ * BK * 4);
  int* toks_ws   = (int*)alloc((size_t)T_ * BK * 4);
  float* stored_ws = (float*)alloc((size_t)T_ * BK * 4);
  int* tpchain   = (int*)alloc((size_t)T_ * BK * 4);

  // one-time: bf16 transposed weights (stay resident in 192MB L2 across steps)
  k_transpose_wout<<<dim3(V_ / 32, H_ / 32), dim3(32, 8), 0, stream>>>(W_out, WoutT);
  k_transpose_wg<<<dim3(KX / 32, G4 / 32), dim3(32, 8), 0, stream>>>(W_ih, W_hh, WgT);
  k_init_state<<<dim3(BK, H_ / 256), 256, 0, stream>>>(h0, c0, h_cur, c_cur);
  k_init_bf_zero<<<(48 * KX + 255) / 256, 256, 0, stream>>>(hbf, xfull);
  k_init_beam<<<1, 64, 0, stream>>>(seqsc, tok_cur);

  float* outs = out;  // d_out[0 .. T*BK*V) holds logps now, permuted in place at the end
  for (int t = 0; t < T_; ++t) {
    k_attn<<<BK, 256, 0, stream>>>(enc, h_cur, ctx);
    k_buildx<<<dim3(BK, KX / 256), 256, 0, stream>>>(embed, tok_cur, ctx, h_cur, xfull);
    // gates: [48,2560] x [2560,4096] -> 256 n-tiles (1 wave each) = 32 blocks
    k_wmma_gemm<<<(G4 / 16) / 8, 256, 0, stream>>>(xfull, WgT, b_lstm, gates, KX, G4);
    k_lstm<<<dim3(BK, H_ / 256), 256, 0, stream>>>(gates, c_cur, h_tmp, c_tmp, hbf);
    // logits: [48,1024] x [1024,32000] -> 2000 n-tiles = 250 blocks
    k_wmma_gemm<<<(V_ / 16) / 8, 256, 0, stream>>>(hbf, WoutT, b_out, logits, H_, V_);
    k_lse<<<BK, 256, 0, stream>>>(logits, rowmax, lse);
    k_logp<<<dim3(BK, V_ / 256), 256, 0, stream>>>(logits, rowmax, lse, outs, t);
    k_topk<<<B_, 512, 0, stream>>>(outs + (size_t)t * BK * V_, seqsc, cand_ws, score_ws);
    k_update<<<1, 64, 0, stream>>>(cand_ws, score_ws, t, seqsc, tok_cur, pred_cur,
                                   preds_ws, toks_ws, stored_ws);
    k_gather_hc<<<dim3(BK, H_ / 256), 256, 0, stream>>>(h_tmp, c_tmp, pred_cur, h_cur, c_cur);
  }

  k_backtrack<<<1, 64, 0, stream>>>(preds_ws, toks_ws, stored_ws, tpchain, out);
  for (int t = 0; t < T_; ++t) {
    hipMemcpyAsync(stage, outs + (size_t)t * BK * V_, (size_t)BK * V_ * 4,
                   hipMemcpyDeviceToDevice, stream);
    k_reorder<<<dim3(BK, V_ / 256), 256, 0, stream>>>(stage, tpchain, outs, t);
  }
  k_writehn<<<dim3(BK, H_ / 256), 256, 0, stream>>>(h_cur, out);
}
